// AMSoftmaxHead_22746146800347
// MI455X (gfx1250) — matrix operand order repaired and checked by reading, not compiled, not run on hardware
//
#include <hip/hip_runtime.h>
#include <hip/hip_bf16.h>
#include <math.h>

// ---------------- types ----------------
typedef __attribute__((ext_vector_type(16))) __bf16          v16bf;
typedef __attribute__((ext_vector_type(8)))  float           v8f;
typedef __attribute__((ext_vector_type(8)))  unsigned short  v8u;
typedef __attribute__((ext_vector_type(4)))  float           v4f;

union Frag { v16bf bf; v8u u[2]; };

#define S_SCALE 30.0f
#define MARGIN  0.4f

// round-to-nearest-even float -> bf16 (as ushort bits)
__device__ __forceinline__ unsigned short f2bf(float x) {
    unsigned int u = __float_as_uint(x);
    unsigned int r = u + 0x7FFFu + ((u >> 16) & 1u);
    return (unsigned short)(r >> 16);
}
__device__ __forceinline__ float bf2f(unsigned short h) {
    return __uint_as_float(((unsigned int)h) << 16);
}

// ---------------- kernel 1: normalize X rows, split to bf16 hi/lo ----------------
__global__ void xnorm_kernel(const float* __restrict__ x,
                             unsigned short* __restrict__ Xhi,
                             unsigned short* __restrict__ Xlo,
                             float* __restrict__ rowSum, int D) {
    const int row = blockIdx.x;
    const float* xr = x + (size_t)row * D;
    float ss = 0.0f;
    for (int d = threadIdx.x; d < D; d += blockDim.x) {
        float v = xr[d];
        ss = fmaf(v, v, ss);
    }
    __shared__ float red[256];
    red[threadIdx.x] = ss;
    __syncthreads();
    for (int s = 128; s > 0; s >>= 1) {
        if ((int)threadIdx.x < s) red[threadIdx.x] += red[threadIdx.x + s];
        __syncthreads();
    }
    const float inv = 1.0f / fmaxf(sqrtf(red[0]), 1e-12f);
    for (int d = threadIdx.x; d < D; d += blockDim.x) {
        float xn = xr[d] * inv;
        unsigned short h = f2bf(xn);
        Xhi[(size_t)row * D + d] = h;
        Xlo[(size_t)row * D + d] = f2bf(xn - bf2f(h));
    }
    if (threadIdx.x == 0) rowSum[row] = 0.0f;
}

// ---------------- kernel 2: W column inverse norms ----------------
__global__ void wnorm_kernel(const float* __restrict__ W,
                             float* __restrict__ invW, int C, int D) {
    const int c = blockIdx.x * blockDim.x + threadIdx.x;
    if (c >= C) return;
    float ss = 0.0f;
    for (int d = 0; d < D; ++d) {
        float v = W[(size_t)d * C + c];
        ss = fmaf(v, v, ss);
    }
    invW[c] = 1.0f / fmaxf(sqrtf(ss), 1e-12f);
}

// ---------------- kernel 3: bf16x3 WMMA GEMM + clip + exp-rowsum ----------------
// Block: 256 threads = 8 waves, tile BM=128 x BN=128, BK=32.
// Wave grid 4(M) x 2(N); each wave: 2x4 tiles of 16x16, 3 WMMAs per tile per K-step.
#define BM 128
#define BN 128
#define BK 32
#define LDSTRIDE 40   // halfwords; 80B rows keep 16B chunks aligned, banks spread

__global__ __launch_bounds__(256, 1)
void amsoftmax_gemm(const unsigned short* __restrict__ Xhi,
                    const unsigned short* __restrict__ Xlo,
                    const float* __restrict__ W,
                    const float* __restrict__ invW,
                    float* __restrict__ cosOut,
                    float* __restrict__ rowSum,
                    int D, int C) {
    __shared__ alignas(16) unsigned short sBhi[BN][LDSTRIDE];
    __shared__ alignas(16) unsigned short sBlo[BN][LDSTRIDE];

    const int tid   = threadIdx.x;
    const int lane  = tid & 31;
    const int wave  = tid >> 5;      // 0..7
    const int waveM = wave & 3;      // 4 waves along M
    const int waveN = wave >> 2;     // 2 waves along N
    const int laneN = lane & 15;
    const int khi   = lane >> 4;     // 0 or 1

    const int mBlock = blockIdx.x * BM;   // 8 blocks along M
    const int nBlock = blockIdx.y * BN;   // 782 blocks along N

    // --- W-tile staging assignment: 4 float4 chunks per thread per K-step ---
    int rowIdx[4], colChunk[4];
    float invc[4][4];
#pragma unroll
    for (int i = 0; i < 4; ++i) {
        int idx = tid + 256 * i;          // 0..1023
        rowIdx[i]   = idx >> 5;           // k row 0..31
        colChunk[i] = (idx & 31) * 4;     // col 0..124
#pragma unroll
        for (int j = 0; j < 4; ++j) {
            int c = nBlock + colChunk[i] + j;
            invc[i][j] = (c < C) ? invW[c] : 0.0f;
        }
    }

    v8f acc[2][4];
#pragma unroll
    for (int mi = 0; mi < 2; ++mi)
#pragma unroll
        for (int ni = 0; ni < 4; ++ni)
            acc[mi][ni] = (v8f){};

    for (int k0 = 0; k0 < D; k0 += BK) {
        // load W tile (f32) to registers
        v4f wv[4];
#pragma unroll
        for (int i = 0; i < 4; ++i) {
            const size_t roff = (size_t)(k0 + rowIdx[i]) * C;
            const int cb = nBlock + colChunk[i];
            if (cb + 3 < C) {
                wv[i] = *(const v4f*)(W + roff + cb);
            } else {
#pragma unroll
                for (int j = 0; j < 4; ++j)
                    wv[i][j] = (cb + j < C) ? W[roff + cb + j] : 0.0f;
            }
        }
        // prefetch next K-tile of W into cache hierarchy
        if (k0 + BK < D) {
            const float* pf = W + (size_t)(k0 + BK + rowIdx[0]) * C + nBlock + colChunk[0];
            __builtin_prefetch(pf, 0, 0);
        }

        __syncthreads();   // previous iteration's consumers done with LDS
        // normalize, split to bf16 hi/lo, store transposed [n][k]
#pragma unroll
        for (int i = 0; i < 4; ++i) {
#pragma unroll
            for (int j = 0; j < 4; ++j) {
                float wn = wv[i][j] * invc[i][j];
                unsigned short h = f2bf(wn);
                unsigned short l = f2bf(wn - bf2f(h));
                sBhi[colChunk[i] + j][rowIdx[i]] = h;
                sBlo[colChunk[i] + j][rowIdx[i]] = l;
            }
        }
        __syncthreads();

        // --- A fragments from global bf16 (ISA 16-bit A 16x32 layout) ---
        Frag ahi[2], alo[2];
#pragma unroll
        for (int mi = 0; mi < 2; ++mi) {
            const size_t row = (size_t)(mBlock + waveM * 32 + mi * 16 + laneN);
            const unsigned short* ph = Xhi + row * D + k0 + khi * 8;
            const unsigned short* pl = Xlo + row * D + k0 + khi * 8;
            ahi[mi].u[0] = *(const v8u*)(ph);
            ahi[mi].u[1] = *(const v8u*)(ph + 16);
            alo[mi].u[0] = *(const v8u*)(pl);
            alo[mi].u[1] = *(const v8u*)(pl + 16);
        }
        // --- B fragments from LDS (dense 16-bit B 32x16: same K split as A) ---
        Frag bhi[4], blo[4];
#pragma unroll
        for (int ni = 0; ni < 4; ++ni) {
            const int cl = waveN * 64 + ni * 16 + laneN;
            const unsigned short* pbh = &sBhi[cl][khi * 8];
            const unsigned short* pbl = &sBlo[cl][khi * 8];
            bhi[ni].u[0] = *(const v8u*)(pbh);
            bhi[ni].u[1] = *(const v8u*)(pbh + 16);
            blo[ni].u[0] = *(const v8u*)(pbl);
            blo[ni].u[1] = *(const v8u*)(pbl + 16);
        }

        // --- bf16x3: hi*hi + hi*lo + lo*hi, f32 accumulate ---
#pragma unroll
        for (int mi = 0; mi < 2; ++mi) {
#pragma unroll
            for (int ni = 0; ni < 4; ++ni) {
                acc[mi][ni] = __builtin_amdgcn_wmma_f32_16x16x32_bf16(
                    false, ahi[mi].bf, false, bhi[ni].bf, (short)0, acc[mi][ni], false, false);
                acc[mi][ni] = __builtin_amdgcn_wmma_f32_16x16x32_bf16(
                    false, ahi[mi].bf, false, blo[ni].bf, (short)0, acc[mi][ni], false, false);
                acc[mi][ni] = __builtin_amdgcn_wmma_f32_16x16x32_bf16(
                    false, alo[mi].bf, false, bhi[ni].bf, (short)0, acc[mi][ni], false, false);
            }
        }
    }

    // ---------------- epilogue: clip, store cossim, exp row-sums ----------------
#pragma unroll
    for (int mi = 0; mi < 2; ++mi) {
        const int mbase = mBlock + waveM * 32 + mi * 16;
        float esum[8];
#pragma unroll
        for (int r = 0; r < 8; ++r) esum[r] = 0.0f;

#pragma unroll
        for (int ni = 0; ni < 4; ++ni) {
            const int c = nBlock + waveN * 64 + ni * 16 + laneN;
            const bool valid = (c < C);
#pragma unroll
            for (int r = 0; r < 8; ++r) {
                float v = acc[mi][ni][r];
                v = fminf(1.0f, fmaxf(-1.0f, v));
                const int m = mbase + khi * 8 + r;   // C/D layout: lanes16-31 hold M=8+r
                if (valid) {
                    cosOut[(size_t)m * C + c] = v;
                    esum[r] += expf(S_SCALE * v);
                }
            }
        }
        // half-wave (16-lane) reduction over the N dimension
#pragma unroll
        for (int r = 0; r < 8; ++r) {
            float s = esum[r];
            s += __shfl_xor(s, 1, 32);
            s += __shfl_xor(s, 2, 32);
            s += __shfl_xor(s, 4, 32);
            s += __shfl_xor(s, 8, 32);
            if (laneN == 0)
                atomicAdd(&rowSum[mbase + khi * 8 + r], s);
        }
    }
}

// ---------------- kernel 4: final loss reduction ----------------
__global__ void loss_kernel(const float* __restrict__ cosOut,
                            const float* __restrict__ rowSum,
                            const int* __restrict__ label,
                            float* __restrict__ outLoss, int B, int C) {
    float acc = 0.0f;
    for (int i = threadIdx.x; i < B; i += blockDim.x) {
        float tgt  = cosOut[(size_t)i * C + label[i]];
        float num  = S_SCALE * (tgt - MARGIN);
        float excl = rowSum[i] - expf(S_SCALE * tgt);
        float L    = num - logf(expf(num) + excl);
        acc += L;
    }
    __shared__ float red[256];
    red[threadIdx.x] = acc;
    __syncthreads();
    for (int s = 128; s > 0; s >>= 1) {
        if ((int)threadIdx.x < s) red[threadIdx.x] += red[threadIdx.x + s];
        __syncthreads();
    }
    if (threadIdx.x == 0) *outLoss = -red[0] / (float)B;
}

// ---------------- host launcher ----------------
extern "C" void kernel_launch(void* const* d_in, const int* in_sizes, int n_in,
                              void* d_out, int out_size, void* d_ws, size_t ws_size,
                              hipStream_t stream) {
    const float* x     = (const float*)d_in[0];
    const float* W     = (const float*)d_in[1];
    const int*   label = (const int*)d_in[2];

    const int B = in_sizes[2];              // 1024
    const int D = in_sizes[0] / B;          // 512
    const int C = in_sizes[1] / D;          // 100000

    float* outCos  = (float*)d_out;
    float* outLoss = outCos + (size_t)B * C;

    char* ws = (char*)d_ws;
    auto al = [](size_t v) { return (v + 255) & ~(size_t)255; };
    size_t off = 0;
    unsigned short* Xhi   = (unsigned short*)(ws + off); off = al(off + (size_t)B * D * 2);
    unsigned short* Xlo   = (unsigned short*)(ws + off); off = al(off + (size_t)B * D * 2);
    float*          invW  = (float*)(ws + off);          off = al(off + (size_t)C * 4);
    float*          rowSum= (float*)(ws + off);          off = al(off + (size_t)B * 4);

    // 1) normalize X rows -> bf16 hi/lo; zero rowSum
    xnorm_kernel<<<B, 256, 0, stream>>>(x, Xhi, Xlo, rowSum, D);

    // 2) W column inverse norms
    wnorm_kernel<<<(C + 255) / 256, 256, 0, stream>>>(W, invW, C, D);

    // 3) WMMA GEMM + clip + exp row-sums.  grid.x = M blocks (adjacent blocks
    //    share the same W tile -> L2 reuse of the 200MB W matrix)
    dim3 grid(B / BM, (C + BN - 1) / BN);
    amsoftmax_gemm<<<grid, 256, 0, stream>>>(Xhi, Xlo, W, invW, outCos, rowSum, D, C);

    // 4) loss
    loss_kernel<<<1, 256, 0, stream>>>(outCos, rowSum, label, outLoss, B, C);
}
